// LinearAttention_34832184771110
// MI455X (gfx1250) — matrix operand order repaired and checked by reading, not compiled, not run on hardware
//
#include <hip/hip_runtime.h>
#include <hip/hip_bf16.h>
#include <math.h>

// ---------------------------------------------------------------------------
// LinearAttention on MI455X (gfx1250, wave32, WMMA)
//   dims: B=8, C=512, L=4096, HEADS=8, DHEAD=64, HID=512, 3*HID=1536
// bf16 WMMA for all GEMMs (fp32 accumulate); async-to-LDS staging with
// double buffering where the layout permits; split-K for the tiny per-head
// context GEMM with a deterministic reduction.
// ---------------------------------------------------------------------------

typedef __attribute__((ext_vector_type(16))) __bf16 v16bf;
typedef __attribute__((ext_vector_type(8)))  float  v8f;
typedef int v4i __attribute__((vector_size(16)));   // matches async builtin

#define BB    8
#define CC    512
#define LL    4096
#define HH    8
#define DH    64
#define HID   512
#define OQKV  1536
#define KSPL  8        // split-K factor for context GEMM

// CDNA5 async-to-LDS path (ASYNCcnt tracked), guarded so the file compiles
// on toolchains that don't expose these builtins.
#if defined(__has_builtin)
#if __has_builtin(__builtin_amdgcn_global_load_async_to_lds_b128) && \
    __has_builtin(__builtin_amdgcn_s_wait_asynccnt)
#define DO_ASYNC 1
#endif
#endif
#ifndef DO_ASYNC
#define DO_ASYNC 0
#endif

#if DO_ASYNC
typedef __attribute__((address_space(1))) v4i* async_gptr;
typedef __attribute__((address_space(3))) v4i* async_lptr;
#endif

static __device__ inline v8f wmma_bf16(v16bf a, v16bf b, v8f c) {
    return __builtin_amdgcn_wmma_f32_16x16x32_bf16(false, a, false, b,
                                                   (short)0, c, false, false);
}

// A-matrix fragment (16x32 bf16): lane L (g=L>>4, r=L&15) holds row r,
// contiguous K chunk [16g, 16g+16) in f[0..15].
static __device__ inline v16bf frag_a(const __bf16* base, int rs, int lane) {
    const int g = lane >> 4, r = lane & 15;
    const __bf16* p = base + r * rs;
    v16bf f;
#pragma unroll
    for (int i = 0; i < 16; ++i) f[i] = p[16 * g + i];
    return f;
}

// B-matrix fragment (32x16 bf16), stored N-major: lane L holds column
// r=L&15, contiguous K chunk [16g, 16g+16).
static __device__ inline v16bf frag_b(const __bf16* base, int rs, int lane) {
    const int g = lane >> 4, r = lane & 15;
    const __bf16* p = base + r * rs + 16 * g;
    v16bf f;
#pragma unroll
    for (int i = 0; i < 16; ++i) f[i] = p[i];
    return f;
}

// ---------------------------------------------------------------------------
// K1: channel RMSNorm of x -> bf16  (xn = x / ||x||_c * g * sqrt(C))
// ---------------------------------------------------------------------------
__global__ __launch_bounds__(256)
void k_rmsnorm_in(const float* __restrict__ x, const float* __restrict__ g,
                  __bf16* __restrict__ xn) {
    const int l = blockIdx.x * 256 + threadIdx.x;
    const int b = blockIdx.y;
    const size_t base = (size_t)b * CC * LL + l;
    float ss = 0.f;
#pragma unroll 4
    for (int c = 0; c < CC; ++c) {
        float v = x[base + (size_t)c * LL];
        ss += v * v;
    }
    float n = fmaxf(sqrtf(ss), 1e-12f);
    float scale = 22.62741699796952f / n;  // sqrt(512)/n
#pragma unroll 4
    for (int c = 0; c < CC; ++c) {
        size_t o = base + (size_t)c * LL;
        xn[o] = (__bf16)(x[o] * scale * g[c]);
    }
}

// ---------------------------------------------------------------------------
// K1b: f32 -> bf16 convert (weights)
// ---------------------------------------------------------------------------
__global__ __launch_bounds__(256)
void k_f32_to_bf16(const float* __restrict__ s, __bf16* __restrict__ d, int n) {
    int i = blockIdx.x * 256 + threadIdx.x;
    if (i < n) d[i] = (__bf16)s[i];
}

// ---------------------------------------------------------------------------
// K2/K6: tiled WMMA GEMM   C[b] = A[M,K] x B[b][K,N] (+ bias)
// block tile 128x128, TK=32, 8 waves, wave tile 32x64 (2x4 accum tiles).
// Double-buffered LDS; A-tile staged with GLOBAL_LOAD_ASYNC_TO_LDS_B128
// when available, B-tile transpose-scattered through registers.
// ---------------------------------------------------------------------------
template <bool BIAS, typename OT>
__global__ __launch_bounds__(256)
void k_gemm_bf16(const __bf16* __restrict__ A, const __bf16* __restrict__ B,
                 OT* __restrict__ C, const float* __restrict__ bias,
                 int M, int N, int K) {
    __shared__ __align__(16) __bf16 As[2][128][32];  // m-major
    __shared__ __align__(16) __bf16 Bs[2][128][32];  // n-major (transposed)

    const int tid  = threadIdx.x;
    const int lane = tid & 31;
    const int wave = tid >> 5;
    const int n0 = blockIdx.x * 128;
    const int m0 = blockIdx.y * 128;
    const int b  = blockIdx.z;

    const __bf16* Bb = B + (size_t)b * K * N;
    OT*           Cb = C + (size_t)b * M * N;

    const int mw = (wave & 3) * 32;
    const int nw = (wave >> 2) * 64;

    // staging assignments
    const int arow = tid >> 1, acol = (tid & 1) * 16;   // A: 32B per thread
    const int bkr  = tid >> 3, bno  = (tid & 7) * 16;   // B: 32B per thread

    v8f acc[2][4];
#pragma unroll
    for (int i = 0; i < 2; ++i)
#pragma unroll
        for (int j = 0; j < 4; ++j)
#pragma unroll
            for (int e = 0; e < 8; ++e) acc[i][j][e] = 0.f;

    auto stage = [&](int buf, int k0) {
        // A tile: straight row-major copy
        const __bf16* asrc = A + (size_t)(m0 + arow) * K + k0 + acol;
#if DO_ASYNC
        __builtin_amdgcn_global_load_async_to_lds_b128(
            (async_gptr)asrc, (async_lptr)&As[buf][arow][acol], 0, 0);
#else
        *(uint4*)&As[buf][arow][acol] = *(const uint4*)asrc;
#endif
        // B tile: coalesced read, transposed scatter into Bs[n][k]
        const __bf16* bsrc = Bb + (size_t)(k0 + bkr) * N + n0 + bno;
        uint4 dv = *(const uint4*)bsrc;
        const __bf16* e = (const __bf16*)&dv;
#pragma unroll
        for (int i = 0; i < 16; ++i) Bs[buf][bno + i][bkr] = e[i];
    };

    stage(0, 0);
#if DO_ASYNC
    __builtin_amdgcn_s_wait_asynccnt(0);
#endif
    __syncthreads();

    int buf = 0;
    for (int k0 = 0; k0 < K; k0 += 32) {
        if (k0 + 32 < K) stage(buf ^ 1, k0 + 32);

        v16bf bfv[4];
#pragma unroll
        for (int nj = 0; nj < 4; ++nj)
            bfv[nj] = frag_b(&Bs[buf][nw + nj * 16][0], 32, lane);
#pragma unroll
        for (int mi = 0; mi < 2; ++mi) {
            v16bf af = frag_a(&As[buf][mw + mi * 16][0], 32, lane);
#pragma unroll
            for (int nj = 0; nj < 4; ++nj)
                acc[mi][nj] = wmma_bf16(af, bfv[nj], acc[mi][nj]);
        }
#if DO_ASYNC
        __builtin_amdgcn_s_wait_asynccnt(0);
#endif
        __syncthreads();
        buf ^= 1;
    }

    // epilogue: D layout -> lane holds N = lane&15, M = r + 8*(lane>>4)
#pragma unroll
    for (int mi = 0; mi < 2; ++mi)
#pragma unroll
        for (int nj = 0; nj < 4; ++nj) {
            const int ncol  = n0 + nw + nj * 16 + (lane & 15);
            const int mbase = m0 + mw + mi * 16 + (lane >> 4) * 8;
#pragma unroll
            for (int r = 0; r < 8; ++r) {
                float v = acc[mi][nj][r];
                if (BIAS) v += bias[mbase + r];
                Cb[(size_t)(mbase + r) * N + ncol] = (OT)v;
            }
        }
}

// ---------------------------------------------------------------------------
// K3a: q softmax over head-dim d (64), *SCALE, in place on bf16 qkv.
// ---------------------------------------------------------------------------
__global__ __launch_bounds__(256)
void k_qsoftmax(__bf16* __restrict__ qkv) {
    const int idx = blockIdx.x * 256 + threadIdx.x;  // B*H*L threads
    const int l  = idx & (LL - 1);
    const int bh = idx >> 12;
    const int b = bh >> 3, h = bh & 7;
    __bf16* base = qkv + ((size_t)b * OQKV + h * DH) * LL + l;
    float mx = -1e30f;
#pragma unroll 8
    for (int d = 0; d < DH; ++d) mx = fmaxf(mx, (float)base[(size_t)d * LL]);
    float s = 0.f;
#pragma unroll 8
    for (int d = 0; d < DH; ++d) s += __expf((float)base[(size_t)d * LL] - mx);
    const float inv = 0.125f / s;  // SCALE = 1/sqrt(64)
#pragma unroll 8
    for (int d = 0; d < DH; ++d) {
        size_t o = (size_t)d * LL;
        base[o] = (__bf16)(__expf((float)base[o] - mx) * inv);
    }
}

// ---------------------------------------------------------------------------
// K3b: k softmax over sequence l (4096), in place. one block per row.
// ---------------------------------------------------------------------------
__global__ __launch_bounds__(256)
void k_ksoftmax(__bf16* __restrict__ qkv) {
    __shared__ float red[256];
    const int row = blockIdx.x;  // B*HID rows
    const int b = row >> 9, c = row & 511;
    __bf16* base = qkv + ((size_t)b * OQKV + HID + c) * LL;
    const int tid = threadIdx.x;

    float mx = -1e30f;
    for (int l = tid; l < LL; l += 256) mx = fmaxf(mx, (float)base[l]);
    red[tid] = mx; __syncthreads();
    for (int s = 128; s > 0; s >>= 1) {
        if (tid < s) red[tid] = fmaxf(red[tid], red[tid + s]);
        __syncthreads();
    }
    mx = red[0]; __syncthreads();

    float sum = 0.f;
    for (int l = tid; l < LL; l += 256) sum += __expf((float)base[l] - mx);
    red[tid] = sum; __syncthreads();
    for (int s = 128; s > 0; s >>= 1) {
        if (tid < s) red[tid] += red[tid + s];
        __syncthreads();
    }
    const float inv = 1.f / red[0];
    for (int l = tid; l < LL; l += 256)
        base[l] = (__bf16)(__expf((float)base[l] - mx) * inv);
}

// ---------------------------------------------------------------------------
// K4: split-K context partials: ctx_part[sp][bh][e][d] (fp32)
//   context[d,e] = sum_n k[d,n] * v[e,n]; per block: one bh, one K-split.
// A rows (k) and B columns (v rows) are K-contiguous -> direct global frags.
// ---------------------------------------------------------------------------
__global__ __launch_bounds__(512)
void k_context(const __bf16* __restrict__ qkv, float* __restrict__ ctx_part) {
    const int bh = blockIdx.x;  // 64
    const int sp = blockIdx.y;  // KSPL splits
    const int b = bh >> 3, h = bh & 7;
    const int lane = threadIdx.x & 31;
    const int wave = threadIdx.x >> 5;  // 16 waves
    const int mi = wave & 3;   // d tile
    const int ni = wave >> 2;  // e tile
    const int g = lane >> 4, r = lane & 15;

    const __bf16* arow = qkv + ((size_t)b * OQKV + HID  + h * DH + mi * 16 + r) * LL;
    const __bf16* brow = qkv + ((size_t)b * OQKV + 1024 + h * DH + ni * 16 + r) * LL;

    v8f acc;
#pragma unroll
    for (int e = 0; e < 8; ++e) acc[e] = 0.f;

    const int nbeg = sp * (LL / KSPL), nend = nbeg + (LL / KSPL);
    for (int n0 = nbeg; n0 < nend; n0 += 32) {
        __builtin_prefetch(arow + n0 + 256, 0, 1);  // -> global_prefetch_b8
        __builtin_prefetch(brow + n0 + 256, 0, 1);
        v16bf af, bf;
        const __bf16* pa = arow + n0;
        const __bf16* pb = brow + n0 + 16 * g;
#pragma unroll
        for (int i = 0; i < 16; ++i) af[i] = pa[16 * g + i];
#pragma unroll
        for (int i = 0; i < 16; ++i) bf[i] = pb[i];
        acc = wmma_bf16(af, bf, acc);
    }

    const int e = ni * 16 + r;           // N index
    const int dbase = mi * 16 + g * 8;   // M index base
    float* out = ctx_part + (((size_t)sp * 64 + bh) * DH + e) * DH;
#pragma unroll
    for (int rr = 0; rr < 8; ++rr) out[dbase + rr] = acc[rr];
}

// K4b: deterministic reduction of split-K partials -> ctxT bf16 [bh][e][d]
__global__ __launch_bounds__(256)
void k_ctx_reduce(const float* __restrict__ part, __bf16* __restrict__ ctxT) {
    const int i = blockIdx.x * 256 + threadIdx.x;  // 64*64*64
    float s = 0.f;
#pragma unroll
    for (int sp = 0; sp < KSPL; ++sp) s += part[(size_t)sp * (64 * DH * DH) + i];
    ctxT[i] = (__bf16)s;
}

// ---------------------------------------------------------------------------
// K5: out[e,n] = sum_d ctxT[e,d] * q[d,n]  per (b,h); M=64, K=64, N=4096.
// q tile staged transposed in LDS (Qs[n][d]); ctxT fragments from global.
// ---------------------------------------------------------------------------
__global__ __launch_bounds__(256)
void k_attnout(const __bf16* __restrict__ qkv, const __bf16* __restrict__ ctxT,
               __bf16* __restrict__ attn) {
    __shared__ __align__(16) __bf16 Qs[128][64];  // [n][d]
    const int n0 = blockIdx.x * 128;
    const int bh = blockIdx.y;
    const int b = bh >> 3, h = bh & 7;
    const int tid = threadIdx.x, lane = tid & 31, wave = tid >> 5;

    // stage q[d, n0..n0+128) transposed: 4 threads/row, 32 elems each
    {
        const int d = tid >> 2, part = (tid & 3) * 32;
        const __bf16* src = qkv + ((size_t)b * OQKV + h * DH + d) * LL + n0 + part;
#pragma unroll
        for (int i = 0; i < 32; ++i) Qs[part + i][d] = src[i];
    }
    __syncthreads();

    const int mi = wave & 3;   // e tile
    const int nh = wave >> 2;  // n half of 64
    const int g = lane >> 4, r = lane & 15;
    const __bf16* arow = ctxT + ((size_t)bh * DH + mi * 16 + r) * DH;

    v8f acc[4];
#pragma unroll
    for (int j = 0; j < 4; ++j)
#pragma unroll
        for (int e = 0; e < 8; ++e) acc[j][e] = 0.f;

#pragma unroll
    for (int k0 = 0; k0 < DH; k0 += 32) {
        v16bf af;
#pragma unroll
        for (int i = 0; i < 16; ++i) af[i] = arow[k0 + 16 * g + i];
#pragma unroll
        for (int nj = 0; nj < 4; ++nj) {
            const __bf16* pb = &Qs[nh * 64 + nj * 16 + r][k0 + 16 * g];
            v16bf bf;
#pragma unroll
            for (int i = 0; i < 16; ++i) bf[i] = pb[i];
            acc[nj] = wmma_bf16(af, bf, acc[nj]);
        }
    }

    __bf16* dst = attn + ((size_t)b * HID + h * DH) * LL;
#pragma unroll
    for (int nj = 0; nj < 4; ++nj) {
        const int n  = n0 + nh * 64 + nj * 16 + r;
        const int mb = mi * 16 + g * 8;
#pragma unroll
        for (int rr = 0; rr < 8; ++rr)
            dst[(size_t)(mb + rr) * LL + n] = (__bf16)acc[nj][rr];
    }
}

// ---------------------------------------------------------------------------
// K7: final channel RMSNorm of o + residual -> d_out (fp32)
// ---------------------------------------------------------------------------
__global__ __launch_bounds__(256)
void k_rmsnorm_out(const float* __restrict__ o, const float* __restrict__ g,
                   const float* __restrict__ x, float* __restrict__ out) {
    const int l = blockIdx.x * 256 + threadIdx.x;
    const int b = blockIdx.y;
    const size_t base = (size_t)b * CC * LL + l;
    float ss = 0.f;
#pragma unroll 4
    for (int c = 0; c < CC; ++c) {
        float v = o[base + (size_t)c * LL];
        ss += v * v;
    }
    float n = fmaxf(sqrtf(ss), 1e-12f);
    float scale = 22.62741699796952f / n;
#pragma unroll 4
    for (int c = 0; c < CC; ++c) {
        size_t off = base + (size_t)c * LL;
        out[off] = o[off] * scale * g[c] + x[off];
    }
}

// ---------------------------------------------------------------------------
extern "C" void kernel_launch(void* const* d_in, const int* in_sizes, int n_in,
                              void* d_out, int out_size, void* d_ws, size_t ws_size,
                              hipStream_t stream) {
    const float* x     = (const float*)d_in[0];
    const float* g_in  = (const float*)d_in[1];
    const float* w_qkv = (const float*)d_in[2];
    const float* w_out = (const float*)d_in[3];
    const float* b_out = (const float*)d_in[4];
    const float* g_out = (const float*)d_in[5];
    float* out = (float*)d_out;

    char* p = (char*)d_ws;
    __bf16* xn     = (__bf16*)p; p += (size_t)BB * CC * LL * 2;          // 32 MiB
    __bf16* wqkv_b = (__bf16*)p; p += (size_t)OQKV * CC * 2;             // 1.5 MiB
    __bf16* wout_b = (__bf16*)p; p += (size_t)CC * HID * 2;              // 0.5 MiB
    __bf16* qkv    = (__bf16*)p; p += (size_t)BB * OQKV * LL * 2;        // 96 MiB
    __bf16* ctxT   = (__bf16*)p; p += (size_t)BB * HH * DH * DH * 2;     // 0.5 MiB
    float*  ctxp   = (float*)p;  p += (size_t)KSPL * 64 * DH * DH * 4;   // 8 MiB
    __bf16* attn   = (__bf16*)p; p += (size_t)BB * HID * LL * 2;         // 32 MiB
    float*  obuf   = (float*)p;  p += (size_t)BB * CC * LL * 4;          // 64 MiB

    // 1. input RMSNorm -> bf16
    k_rmsnorm_in<<<dim3(LL / 256, BB), 256, 0, stream>>>(x, g_in, xn);
    // 1b. weights -> bf16
    k_f32_to_bf16<<<(OQKV * CC + 255) / 256, 256, 0, stream>>>(w_qkv, wqkv_b, OQKV * CC);
    k_f32_to_bf16<<<(CC * HID + 255) / 256, 256, 0, stream>>>(w_out, wout_b, CC * HID);
    // 2. QKV projection: [1536,512] x [512,4096] per batch
    k_gemm_bf16<false, __bf16><<<dim3(LL / 128, OQKV / 128, BB), 256, 0, stream>>>(
        wqkv_b, xn, qkv, nullptr, OQKV, LL, CC);
    // 3. softmaxes (in place on qkv)
    k_qsoftmax<<<(BB * HH * LL) / 256, 256, 0, stream>>>(qkv);
    k_ksoftmax<<<BB * HID, 256, 0, stream>>>(qkv);
    // 4. per-head context = k v^T, split-K partials + deterministic reduce
    k_context<<<dim3(BB * HH, KSPL), 512, 0, stream>>>(qkv, ctxp);
    k_ctx_reduce<<<(64 * DH * DH) / 256, 256, 0, stream>>>(ctxp, ctxT);
    // 5. per-head out = ctx^T q
    k_attnout<<<dim3(LL / 128, BB * HH), 256, 0, stream>>>(qkv, ctxT, attn);
    // 6. output projection + bias
    k_gemm_bf16<true, float><<<dim3(LL / 128, CC / 128, BB), 256, 0, stream>>>(
        wout_b, attn, obuf, b_out, CC, LL, HID);
    // 7. output RMSNorm + residual
    k_rmsnorm_out<<<dim3(LL / 256, BB), 256, 0, stream>>>(obuf, g_out, x, out);
}
